// RNN_22806276341782
// MI455X (gfx1250) — matrix operand order repaired and checked by reading, not compiled, not run on hardware
//
#include <hip/hip_runtime.h>
#include <math.h>

typedef __attribute__((ext_vector_type(16))) _Float16 v16h;
typedef __attribute__((ext_vector_type(8)))  float    v8f;
typedef __attribute__((ext_vector_type(2)))  float    v2f;

#define GB 256
#define GT 2048
#define GH 64
// row stride (in halves) of the shared h tile; multiple of 8 so every
// 8-half chunk is 16B aligned -> ds_load_b128
#define HSTRIDE 72

static __device__ inline v8f wmma_h(v16h a, v16h b, v8f c) {
  // D = A(16x32 f16) * B(32x16 f16) + C(16x16 f32)
  return __builtin_amdgcn_wmma_f32_16x16x32_f16(false, a, false, b, (short)0, c,
                                                false, false);
}
static __device__ inline v8f wmma_s(v2f a, v2f b, v8f c) {
  // D = A(16x4 f32) * B(4x16 f32) + C(16x16 f32)
  return __builtin_amdgcn_wmma_f32_16x16x4_f32(false, a, false, b, (short)0, c,
                                               false, false);
}

// fast sigmoid: v_exp_f32 + v_rcp_f32 (trans pipe, co-executes with WMMA)
static __device__ inline float fast_sigmoid(float v) {
  return __builtin_amdgcn_rcpf(1.0f + __expf(-v));
}
// fast tanh: hardware v_tanh_f32 if available, else exp/rcp identity
static __device__ inline float fast_tanh(float v) {
#if __has_builtin(__builtin_amdgcn_tanhf)
  return __builtin_amdgcn_tanhf(v);
#else
  return 1.0f - 2.0f * __builtin_amdgcn_rcpf(1.0f + __expf(2.0f * v));
#endif
}

__global__ __launch_bounds__(32, 1)
void gru_fused(const float* __restrict__ x,   // [B, T, 2]
               const float* __restrict__ Wih, // [192, 2]
               const float* __restrict__ Whh, // [192, 64]
               const float* __restrict__ bih, // [192]
               const float* __restrict__ bhh, // [192]
               float* __restrict__ out)       // [B, 64]
{
  const int lane = threadIdx.x & 31;
  const int nn   = lane & 15;   // column within 16-wide tile / A-row
  const int hi   = lane >> 4;   // lane half (0/1)
  const int bt   = blockIdx.x;  // batch tile: rows [16*bt, 16*bt+16)

  __shared__ __align__(16) _Float16 hS[16 * HSTRIDE];

  // ------------------------------------------------------------------
  // Build loop-invariant WMMA B operands, register-resident.
  //
  // f16 B tiles (32x16, K = h-dim): B[k][n] = Whh[gate_col][k].
  // Lane layout (dense 16x16x32 B): lane holds column n = lane%16;
  // half j (0..15) holds K = 16*(lane/16) + j within the 32-K tile.
  // ------------------------------------------------------------------
  v16h Bh[12][2];
#pragma unroll
  for (int ct = 0; ct < 12; ++ct) {
#pragma unroll
    for (int kt = 0; kt < 2; ++kt) {
      const int c = ct * 16 + nn; // gate row 0..191 (r:0-63, z:64-127, n:128-191)
      v16h b;
#pragma unroll
      for (int i = 0; i < 16; ++i) {
        const int k = kt * 32 + hi * 16 + i;
        b[i] = (_Float16)Whh[c * GH + k];
      }
      Bh[ct][kt] = b;
    }
  }

  // f32 B tiles (4x16) for the input projection + bias fusion.
  // A rows are [x0, x1, 1, 0]; so B rows are [Wih[:,0], Wih[:,1], bias, 0].
  // Tiles 0..7  : r|z fused pre-activation, bias = b_ih + b_hh
  // Tiles 8..11 : hg_n term, only bias row (b_hh_n)
  // Tiles 12..15: xg_n term, Wih rows + b_ih_n
  // Lane layout (4x16 f32 B): lane holds col n = lane%16, K = 2*(lane/16)+{0,1}.
  v2f Bs[16];
#pragma unroll
  for (int tI = 0; tI < 16; ++tI) {
    const int g = (tI < 12) ? (tI * 16 + nn) : ((tI - 4) * 16 + nn);
    float r0, r1, r2;
    if (tI < 8)       { r0 = Wih[g * 2]; r1 = Wih[g * 2 + 1]; r2 = bih[g] + bhh[g]; }
    else if (tI < 12) { r0 = 0.0f;       r1 = 0.0f;           r2 = bhh[g]; }
    else              { r0 = Wih[g * 2]; r1 = Wih[g * 2 + 1]; r2 = bih[g]; }
    v2f lo = {r0, r1};
    v2f hiv = {r2, 0.0f};
    Bs[tI] = hi ? hiv : lo;
  }

  // ------------------------------------------------------------------
  // Recurrent state: h in f32 C-layout (4 tiles of 16x16, VGPR g holds
  // row g + 8*hi, col = nn) and as f16 A operands (2 K-tiles of 16x32).
  // ------------------------------------------------------------------
  v8f  hF[4];
#pragma unroll
  for (int j = 0; j < 4; ++j) hF[j] = v8f{};
  v16h A0 = v16h{}, A1 = v16h{};

  // x double buffer; all 32 lanes load (lane&15 row -> lane pairs coalesce),
  // avoiding a per-step exec-mask branch. Only lanes<16 feed the A operand.
  const float* xrow = x + (size_t)(bt * 16 + nn) * GT * 2;
  float2 xc = *(const float2*)(xrow);

#pragma unroll 1
  for (int t = 0; t < GT; ++t) {
    // prefetch next step's x (clamped; last-iteration value is unused)
    const int tn = (t + 1 < GT) ? (t + 1) : t;
    const float2 xn = *(const float2*)(xrow + (size_t)tn * 2);

    // A operand for the f32 WMMA: rows [x0, x1, 1, 0]
    v2f a2lo = {xc.x, xc.y};
    v2f a2hi = {1.0f, 0.0f};
    v2f A2   = hi ? a2hi : a2lo;

#pragma unroll
    for (int j = 0; j < 4; ++j) {
      // r gate pre-activation (tile j), fully fused incl. both biases
      v8f ar = v8f{};
      ar = wmma_h(A0, Bh[j][0], ar);
      ar = wmma_h(A1, Bh[j][1], ar);
      ar = wmma_s(A2, Bs[j], ar);
      // z gate pre-activation (tile 4+j)
      v8f az = v8f{};
      az = wmma_h(A0, Bh[4 + j][0], az);
      az = wmma_h(A1, Bh[4 + j][1], az);
      az = wmma_s(A2, Bs[4 + j], az);
      // hg_n = h*Whh_n + b_hh_n (tile 8+j)
      v8f ag = v8f{};
      ag = wmma_h(A0, Bh[8 + j][0], ag);
      ag = wmma_h(A1, Bh[8 + j][1], ag);
      ag = wmma_s(A2, Bs[8 + j], ag);
      // xg_n = x*Wih_n + b_ih_n (tile 12+j)
      v8f ax = v8f{};
      ax = wmma_s(A2, Bs[12 + j], ax);

      // elementwise gate math in C layout (8 rows per lane)
      v8f hOld = hF[j];
      v8f hNew;
#pragma unroll
      for (int g = 0; g < 8; ++g) {
        const float r = fast_sigmoid(ar[g]);
        const float z = fast_sigmoid(az[g]);
        const float n = fast_tanh(fmaf(r, ag[g], ax[g]));
        hNew[g] = fmaf(z, hOld[g] - n, n); // (1-z)*n + z*h
      }
      hF[j] = hNew;

      // stage h_new to LDS (row-major f16) for the C->A relayout
      const int col = j * 16 + nn;
#pragma unroll
      for (int g = 0; g < 8; ++g) {
        const int row = g + hi * 8;
        hS[row * HSTRIDE + col] = (_Float16)hNew[g];
      }
    }

    __syncthreads(); // single wave: lowers to ds-count wait

    // Reload A operands in the same contiguous K order as Bh:
    // lane m%16 holds row m; halves 0..7 = K chunk [kt*32 + 16*hi ..],
    // halves 8..15 = K chunk [kt*32 + 16*hi + 8 ..]  (each 16B -> b128)
#pragma unroll
    for (int kt = 0; kt < 2; ++kt) {
      const int ccb = kt * 32 + hi * 16;
      const uint4 q0 = *(const uint4*)(&hS[nn * HSTRIDE + ccb]);
      const uint4 q1 = *(const uint4*)(&hS[nn * HSTRIDE + ccb + 8]);
      union { uint4 q[2]; v16h v; } u;
      u.q[0] = q0;
      u.q[1] = q1;
      if (kt == 0) A0 = u.v; else A1 = u.v;
    }

    xc = xn;
  }

  // write final hidden state [B, 64]
#pragma unroll
  for (int j = 0; j < 4; ++j) {
#pragma unroll
    for (int g = 0; g < 8; ++g) {
      const int row = g + hi * 8;
      const int col = j * 16 + nn;
      out[(size_t)(bt * 16 + row) * GH + col] = hF[j][g];
    }
  }
}

extern "C" void kernel_launch(void* const* d_in, const int* in_sizes, int n_in,
                              void* d_out, int out_size, void* d_ws, size_t ws_size,
                              hipStream_t stream) {
  (void)in_sizes; (void)n_in; (void)d_ws; (void)ws_size; (void)out_size;
  const float* x   = (const float*)d_in[0];
  const float* Wih = (const float*)d_in[1];
  const float* Whh = (const float*)d_in[2];
  const float* bih = (const float*)d_in[3];
  const float* bhh = (const float*)d_in[4];
  float* out = (float*)d_out;

  gru_fused<<<dim3(GB / 16), dim3(32), 0, stream>>>(x, Wih, Whh, bih, bhh, out);
}
